// HGTLayer_17592186044972
// MI455X (gfx1250) — compile-verified, matrix operand order checked by reading, not run on hardware
//
#include <hip/hip_runtime.h>
#include <hip/hip_bf16.h>

// HGT layer for gfx1250: WMMA f16 GEMMs (B staged in LDS, async global->LDS
// when available) + fused block-diagonal relation transforms + VALU/atomic
// edge softmax & aggregation.
//
// Sizes (fixed by reference): N=65536 nodes/type, IN=OUT=256, H=8, DK=32,
// E=262144 edges/relation.

#define NN 65536
#define EE 262144

typedef __attribute__((ext_vector_type(16))) _Float16 v16h;
typedef __attribute__((ext_vector_type(8)))  float    v8f;
typedef int v4i __attribute__((vector_size(16)));
typedef __attribute__((address_space(1))) v4i* gptr_v4i;   // global int4*
typedef __attribute__((address_space(3))) v4i* lptr_v4i;   // LDS int4*

#if __has_builtin(__builtin_amdgcn_global_load_async_to_lds_b128) && \
    __has_builtin(__builtin_amdgcn_s_wait_asynccnt)
#define USE_ASYNC_LDS 1
#else
#define USE_ASYNC_LDS 0
#endif

// ---------------------------------------------------------------- fills
__global__ void fill_f32(float* __restrict__ p, float v, int n) {
  int i = blockIdx.x * blockDim.x + threadIdx.x;
  if (i < n) p[i] = v;
}
__global__ void fill_i32(int* __restrict__ p, int v, int n) {
  int i = blockIdx.x * blockDim.x + threadIdx.x;
  if (i < n) p[i] = v;
}

// --------------------------------------------- fuse W @ blockdiag(R), b @ blockdiag(R)
// W: [256,256] row-major (in_dim x out_dim), b: [256], R: [H=8][32][32].
// Wf[k, h*32+e] = sum_d W[k, h*32+d] * R[h][d][e]   (and same for bias)
__global__ void fuse_weight(const float* __restrict__ W, const float* __restrict__ b,
                            const float* __restrict__ R,
                            float* __restrict__ Wf, float* __restrict__ bf) {
  int t = blockIdx.x * blockDim.x + threadIdx.x;   // 0 .. 65535
  int n = t & 255;
  int k = t >> 8;
  int h = n >> 5;
  int dc = n & 31;
  const float* Rh = R + h * 32 * 32;
  const float* Wr = W + k * 256 + h * 32;
  float acc = 0.f;
#pragma unroll
  for (int d = 0; d < 32; ++d) acc += Wr[d] * Rh[d * 32 + dc];
  Wf[t] = acc;
  if (k == 0) {
    const float* br = b + h * 32;
    float accb = 0.f;
#pragma unroll
    for (int d = 0; d < 32; ++d) accb += br[d] * Rh[d * 32 + dc];
    bf[n] = accb;
  }
}

// --------------------------------------------- swizzle f32 [256,256] -> f16 WMMA B-fragments
// Bfrag layout: [kt:8][nt:16][lane:32][e:16] f16 so each lane reads 32 contiguous bytes.
// lane holds column n = nt*16 + (lane&15); half = lane>>4;
// element e -> k = kt*32 + (e<8 ? half*8+e : 16+half*8+(e-8))   (mirrors 16-bit A layout)
__global__ void swizzleB(const float* __restrict__ W, _Float16* __restrict__ Bfrag) {
  int t = blockIdx.x * blockDim.x + threadIdx.x;   // 0 .. 65535
  int e    = t & 15;
  int lane = (t >> 4) & 31;
  int nt   = (t >> 9) & 15;
  int kt   = t >> 13;
  int half = lane >> 4;
  int n = nt * 16 + (lane & 15);
  int klocal = (e < 8) ? (half * 8 + e) : (16 + half * 8 + (e - 8));
  int k = kt * 32 + klocal;
  Bfrag[t] = (_Float16)W[k * 256 + n];
}

// --------------------------------------------- A-fragment loader (16-bit A 16x32 layout)
__device__ __forceinline__ v16h load_afrag(const float* __restrict__ A, int row,
                                           int kt, int half) {
  const float* arow = A + (size_t)row * 256 + kt * 32 + half * 8;
  const float4* a4 = (const float4*)arow;
  float4 x0 = a4[0];          // k+0..3
  float4 x1 = a4[1];          // k+4..7
  float4 x2 = a4[4];          // k+16..19
  float4 x3 = a4[5];          // k+20..23
  v16h af;
  af[0]=(_Float16)x0.x;  af[1]=(_Float16)x0.y;  af[2]=(_Float16)x0.z;  af[3]=(_Float16)x0.w;
  af[4]=(_Float16)x1.x;  af[5]=(_Float16)x1.y;  af[6]=(_Float16)x1.z;  af[7]=(_Float16)x1.w;
  af[8]=(_Float16)x2.x;  af[9]=(_Float16)x2.y;  af[10]=(_Float16)x2.z; af[11]=(_Float16)x2.w;
  af[12]=(_Float16)x3.x; af[13]=(_Float16)x3.y; af[14]=(_Float16)x3.z; af[15]=(_Float16)x3.w;
  return af;
}

// --------------------------------------------- WMMA GEMM: C[M,256] = scaleA*(A[M,256]@B) + bias
// Optional epilogue: C = C*sigmoid(skip[skipIdx]) + resid*(1-sigmoid(...))
// Whole swizzled B (128KB f16) is staged in LDS once per block and shared by
// all 8 waves. One wave computes a 32-row strip (two 16-row m-tiles) across
// all 256 output columns, so each B fragment read from LDS feeds 2 WMMAs.
__global__ __launch_bounds__(256)
void gemm_rowstrip(const float* __restrict__ A, const _Float16* __restrict__ Bfrag,
                   const float* __restrict__ bias, float* __restrict__ C, int M,
                   float scaleA, const float* __restrict__ skip, int skipIdx,
                   const float* __restrict__ resid) {
  extern __shared__ _Float16 Bl[];   // 65536 halves = 128 KB

  // ---- cooperative stage of B fragments into LDS (async path on CDNA5) ----
  {
    int t = threadIdx.x;
#if USE_ASYNC_LDS
#pragma unroll 4
    for (int i = 0; i < 32; ++i) {
      int off16 = (i * 256 + t) * 8;  // element offset; 16B chunks
      __builtin_amdgcn_global_load_async_to_lds_b128(
          (gptr_v4i)(Bfrag + off16),
          (lptr_v4i)(Bl + off16), 0, 0);
    }
    __builtin_amdgcn_s_wait_asynccnt(0);
#else
    const uint4* gs = (const uint4*)Bfrag;
    uint4* ls = (uint4*)Bl;
#pragma unroll 4
    for (int i = 0; i < 32; ++i) ls[i * 256 + t] = gs[i * 256 + t];
#endif
  }
  __syncthreads();

  int wave = (int)((blockIdx.x * blockDim.x + threadIdx.x) >> 5);
  int lane = threadIdx.x & 31;
  int m0 = wave * 32;
  if (m0 >= M) return;               // wave-uniform: EXEC all-ones for WMMA
  int half = lane >> 4;
  int r0 = m0 + (lane & 15);
  int r1 = r0 + 16;

  v8f acc0[16], acc1[16];
#pragma unroll
  for (int i = 0; i < 16; ++i) {
    acc0[i] = (v8f){0.f,0.f,0.f,0.f,0.f,0.f,0.f,0.f};
    acc1[i] = (v8f){0.f,0.f,0.f,0.f,0.f,0.f,0.f,0.f};
  }

  for (int kt = 0; kt < 8; ++kt) {
    v16h a0 = load_afrag(A, r0, kt, half);
    v16h a1 = load_afrag(A, r1, kt, half);
    const _Float16* bp = Bl + kt * (16 * 32 * 16) + lane * 16;
#pragma unroll
    for (int nt = 0; nt < 16; ++nt) {
      v16h b = *(const v16h*)(bp + nt * (32 * 16));
      acc0[nt] = __builtin_amdgcn_wmma_f32_16x16x32_f16(
          false, a0, false, b, (short)0, acc0[nt], false, false);
      acc1[nt] = __builtin_amdgcn_wmma_f32_16x16x32_f16(
          false, a1, false, b, (short)0, acc1[nt], false, false);
    }
  }

  float alpha = 1.f, beta = 0.f;
  if (skip) {
    float s = skip[skipIdx];
    alpha = 1.f / (1.f + __expf(-s));
    beta = 1.f - alpha;
  }
  int col0 = lane & 15;
#pragma unroll
  for (int mt = 0; mt < 2; ++mt) {
    int mb = m0 + mt * 16 + half * 8;   // C/D layout: VGPR r -> M = r + 8*half
    const v8f* acc = mt ? acc1 : acc0;
#pragma unroll
    for (int nt = 0; nt < 16; ++nt) {
      int col = nt * 16 + col0;
      float b = bias ? bias[col] : 0.f;
#pragma unroll
      for (int r = 0; r < 8; ++r) {
        size_t idx = (size_t)(mb + r) * 256 + col;
        float val = acc[nt][r] * scaleA + b;
        if (resid) val = val * alpha + resid[idx] * beta;
        C[idx] = val;
      }
    }
  }
}

// --------------------------------------------- edge phase
// monotonic int encoding of float for atomicMax
__device__ __forceinline__ int enc_f32(float f) {
  int i = __float_as_int(f);
  return (i >= 0) ? i : (i ^ 0x7fffffff);
}
__device__ __forceinline__ float dec_f32(int i) {
  return __int_as_float((i >= 0) ? i : (i ^ 0x7fffffff));
}

__global__ void edge_score(const float* __restrict__ Q, const float* __restrict__ K,
                           const int* __restrict__ src, const int* __restrict__ dst,
                           const float* __restrict__ pri, float* __restrict__ score,
                           int* __restrict__ mEnc, int ne) {
  int t = blockIdx.x * blockDim.x + threadIdx.x;
  if (t >= ne * 8) return;
  int h = t & 7;
  int e = t >> 3;
  int s = src[e], d = dst[e];
  const float4* qp = (const float4*)(Q + (size_t)d * 256 + h * 32);
  const float4* kp = (const float4*)(K + (size_t)s * 256 + h * 32);
  float acc = 0.f;
#pragma unroll
  for (int i = 0; i < 8; ++i) {
    float4 qv = qp[i], kv = kp[i];
    acc += qv.x * kv.x + qv.y * kv.y + qv.z * kv.z + qv.w * kv.w;
  }
  float sc = acc * pri[h] * 0.17677669529663687f;   // 1/sqrt(32)
  score[t] = sc;
  atomicMax(&mEnc[d * 8 + h], enc_f32(sc));
}

__global__ void edge_exp(const int* __restrict__ dst, const int* __restrict__ mEnc,
                         float* __restrict__ score, float* __restrict__ den, int ne) {
  int t = blockIdx.x * blockDim.x + threadIdx.x;
  if (t >= ne * 8) return;
  int h = t & 7;
  int e = t >> 3;
  int d = dst[e];
  int enc = mEnc[d * 8 + h];
  float m = (enc == (int)0x80000000) ? 0.f : dec_f32(enc);
  float ex = __expf(score[t] - m);
  score[t] = ex;                       // in-place: score -> exp
  atomicAdd(&den[d * 8 + h], ex);
}

__global__ void edge_agg(const int* __restrict__ src, const int* __restrict__ dst,
                         const float* __restrict__ V, const float* __restrict__ ex,
                         const float* __restrict__ den, float* __restrict__ agg, int ne) {
  int t = blockIdx.x * blockDim.x + threadIdx.x;
  if (t >= ne * 8) return;
  int h = t & 7;
  int e = t >> 3;
  int s = src[e], d = dst[e];
  float dn = den[d * 8 + h];
  float a = ex[t] / (dn > 0.f ? dn : 1.f);
  const float* vp = V + (size_t)s * 256 + h * 32;
  float* ap = agg + (size_t)d * 256 + h * 32;
#pragma unroll
  for (int i = 0; i < 32; ++i) atomicAdd(&ap[i], vp[i] * a);
}

// ---------------------------------------------------------------- launch
extern "C" void kernel_launch(void* const* d_in, const int* in_sizes, int n_in,
                              void* d_out, int out_size, void* d_ws, size_t ws_size,
                              hipStream_t stream) {
  const float* h_p = (const float*)d_in[0];
  const float* h_a = (const float*)d_in[1];
  const int* esrc[3] = {(const int*)d_in[2], (const int*)d_in[4], (const int*)d_in[6]};
  const int* edst[3] = {(const int*)d_in[3], (const int*)d_in[5], (const int*)d_in[7]};
  const float* Wk = (const float*)d_in[8];
  const float* bk = (const float*)d_in[9];
  const float* Wv = (const float*)d_in[10];
  const float* bv = (const float*)d_in[11];
  const float* Wq = (const float*)d_in[12];
  const float* bq = (const float*)d_in[13];
  const float* Wa = (const float*)d_in[14];
  const float* ba = (const float*)d_in[15];
  const float* rel_pri = (const float*)d_in[16];
  const float* rel_att = (const float*)d_in[17];
  const float* rel_msg = (const float*)d_in[18];
  const float* skip = (const float*)d_in[19];
  float* out = (float*)d_out;

  char* ws = (char*)d_ws;
  size_t off = 0;
  auto alloc = [&](size_t bytes) -> char* {
    char* p = ws + off;
    off += (bytes + 255) & ~(size_t)255;
    return p;
  };
  const size_t NODE_F = (size_t)NN * 256;
  float*     Qp    = (float*)alloc(NODE_F * 4);
  float*     Qa    = (float*)alloc(NODE_F * 4);
  float*     Kbuf  = (float*)alloc(NODE_F * 4);
  float*     Vbuf  = (float*)alloc(NODE_F * 4);
  float*     aggP  = (float*)alloc(NODE_F * 4);
  float*     aggA  = (float*)alloc(NODE_F * 4);
  float*     score = (float*)alloc((size_t)EE * 8 * 4);
  int*       mEnc  = (int*)  alloc((size_t)NN * 8 * 4);
  float*     den   = (float*)alloc((size_t)NN * 8 * 4);
  float*     Wf    = (float*)alloc(256 * 256 * 4);
  float*     bf    = (float*)alloc(256 * 4);
  _Float16*  frag  = (_Float16*)alloc(8 * 16 * 32 * 16 * 2);

  const int THR = 256;
  const int LDS_BYTES = 8 * 16 * 32 * 16 * 2;  // 128 KB dynamic LDS for B
  dim3 blk(THR);
  dim3 gFill((int)(NODE_F / THR));           // 65536 blocks
  dim3 gNH((NN * 8) / THR);                  // 2048
  dim3 gMat((256 * 256) / THR);              // 256
  dim3 gEdge((EE * 8) / THR);                // 8192
  dim3 gGemm((NN / 32) / 8);                 // 256 blocks * 8 waves = 2048 strips

  // zero cross-relation accumulators
  fill_f32<<<gFill, blk, 0, stream>>>(aggP, 0.f, (int)NODE_F);
  fill_f32<<<gFill, blk, 0, stream>>>(aggA, 0.f, (int)NODE_F);

  // Q projections (no relation transform): Qp = h_p@Wq0+bq0 ; Qa = h_a@Wq1+bq1
  swizzleB<<<gMat, blk, 0, stream>>>(Wq, frag);
  gemm_rowstrip<<<gGemm, blk, LDS_BYTES, stream>>>(h_p, frag, bq, Qp, NN, 1.f, nullptr, 0, nullptr);
  swizzleB<<<gMat, blk, 0, stream>>>(Wq + 256 * 256, frag);
  gemm_rowstrip<<<gGemm, blk, LDS_BYTES, stream>>>(h_a, frag, bq + 256, Qa, NN, 1.f, nullptr, 0, nullptr);

  // relations: 0 author->paper, 1 paper->paper, 2 paper->author
  const int srcType[3] = {1, 0, 0};
  const float* srcH[3] = {h_a, h_p, h_p};
  const float* dstQ[3] = {Qp, Qp, Qa};
  float* aggT[3] = {aggP, aggP, aggA};

  for (int r = 0; r < 3; ++r) {
    int s = srcType[r];
    fill_i32<<<gNH, blk, 0, stream>>>(mEnc, (int)0x80000000, NN * 8);
    fill_f32<<<gNH, blk, 0, stream>>>(den, 0.f, NN * 8);

    // K = h_src @ (Wk[s] @ blockdiag(att[r])) + bk[s] @ blockdiag(att[r])
    fuse_weight<<<gMat, blk, 0, stream>>>(Wk + (size_t)s * 256 * 256, bk + s * 256,
                                          rel_att + (size_t)r * 8 * 32 * 32, Wf, bf);
    swizzleB<<<gMat, blk, 0, stream>>>(Wf, frag);
    gemm_rowstrip<<<gGemm, blk, LDS_BYTES, stream>>>(srcH[r], frag, bf, Kbuf, NN, 1.f, nullptr, 0, nullptr);

    // V = h_src @ (Wv[s] @ blockdiag(msg[r])) + bv[s] @ blockdiag(msg[r])
    fuse_weight<<<gMat, blk, 0, stream>>>(Wv + (size_t)s * 256 * 256, bv + s * 256,
                                          rel_msg + (size_t)r * 8 * 32 * 32, Wf, bf);
    swizzleB<<<gMat, blk, 0, stream>>>(Wf, frag);
    gemm_rowstrip<<<gGemm, blk, LDS_BYTES, stream>>>(srcH[r], frag, bf, Vbuf, NN, 1.f, nullptr, 0, nullptr);

    // edge softmax + aggregation
    edge_score<<<gEdge, blk, 0, stream>>>(dstQ[r], Kbuf, esrc[r], edst[r],
                                          rel_pri + r * 8, score, mEnc, EE);
    edge_exp<<<gEdge, blk, 0, stream>>>(edst[r], mEnc, score, den, EE);
    edge_agg<<<gEdge, blk, 0, stream>>>(esrc[r], edst[r], Vbuf, score, den, aggT[r], EE);
  }

  // outputs: paper gets mean of 2 relations (scaleA=0.5), author one relation
  swizzleB<<<gMat, blk, 0, stream>>>(Wa, frag);
  gemm_rowstrip<<<gGemm, blk, LDS_BYTES, stream>>>(aggP, frag, ba, out, NN, 0.5f, skip, 0, h_p);
  swizzleB<<<gMat, blk, 0, stream>>>(Wa + 256 * 256, frag);
  gemm_rowstrip<<<gGemm, blk, LDS_BYTES, stream>>>(aggA, frag, ba + 256, out + NODE_F, NN, 1.f, skip, 1, h_a);
}
